// SDPADilatedAttention_30331059044984
// MI455X (gfx1250) — matrix operand order repaired and checked by reading, not compiled, not run on hardware
//
#include <hip/hip_runtime.h>

// DILATION=1 => block-diagonal attention over 256-token segments.
// f16 data / f32-accumulate WMMA pipeline:
//   cvt(x,W) -> QKV GEMMs (v_wmma_f32_16x16x32_f16, f16 out) ->
//   per-segment attention (f16 WMMA, exact f32 softmax, f16 ctx) ->
//   output projection (f16 WMMA, f32 out + bias)

#define EMBED 768
#define NH    12
#define HD    64
#define SEQ   2048
#define BATCH 2
#define SEG   256
#define NSEG  (SEQ / SEG)
#define MTOT  (BATCH * SEQ)

#define ASTR 72    // LDS row stride (halfs) for GEMM A/B tiles (144B: 16B-aligned, conflict-free frag reads)
#define KSTR 72    // LDS row stride for K segment
#define VSTR 264   // LDS row stride for transposed V (64 x 256)
#define SSTR 40    // LDS row stride for P staging tiles (16 x 32)

typedef _Float16 half_t;
typedef __attribute__((ext_vector_type(4)))  _Float16 v4h;
typedef __attribute__((ext_vector_type(8)))  _Float16 v8h;
typedef __attribute__((ext_vector_type(16))) _Float16 v16h;
typedef __attribute__((ext_vector_type(8)))  float    v8f;

union Frag16 {
  v16h v;
  struct { v8h lo, hi; } p;
};

__device__ __forceinline__ v8f wmma32(v16h a, v16h b, v8f c) {
  // D(16x16,f32) = A(16x32,f16) * B(32x16,f16) + C
  return __builtin_amdgcn_wmma_f32_16x16x32_f16(false, a, false, b, (short)0, c,
                                                false, false);
}

// ---------------------------------------------------------------------------
// f32 -> f16 conversion (n4 = element count / 4; all sizes divisible by 1024)
// ---------------------------------------------------------------------------
__global__ __launch_bounds__(256) void cvt_f32_f16(const float* __restrict__ s,
                                                   half_t* __restrict__ d,
                                                   int n4) {
  const int i = blockIdx.x * 256 + threadIdx.x;
  if (i >= n4) return;
  const float4 f = ((const float4*)s)[i];
  v4h h;
  h[0] = (half_t)f.x; h[1] = (half_t)f.y;
  h[2] = (half_t)f.z; h[3] = (half_t)f.w;
  ((v4h*)d)[i] = h;
}

// ---------------------------------------------------------------------------
// GEMM: D[m][n] = sum_k X[m][k] * W[n][k] + bias[n]   (X,W in f16)
// mode 0: f16 scatter into (B,H,S,D) (QKV);  mode 1: f32 row-major [M][768].
// grid=(N/128, M/128, nz), block=256 (8 waves): wave = 32x64 of a 128x128 tile
// ---------------------------------------------------------------------------
__global__ __launch_bounds__(256) void gemm_f16_kernel(
    const half_t* __restrict__ X, const half_t* __restrict__ W0,
    const half_t* __restrict__ W1, const half_t* __restrict__ W2,
    const float* __restrict__ bias0, const float* __restrict__ bias1,
    const float* __restrict__ bias2, half_t* __restrict__ D0,
    half_t* __restrict__ D1, half_t* __restrict__ D2, float* __restrict__ Df,
    int mode) {
  __shared__ half_t lA[128 * ASTR];
  __shared__ half_t lB[128 * ASTR];

  const int z = blockIdx.z;
  const half_t* W   = (z == 0) ? W0 : (z == 1) ? W1 : W2;
  const float* bias = (z == 0) ? bias0 : (z == 1) ? bias1 : bias2;
  half_t* Dh        = (z == 0) ? D0 : (z == 1) ? D1 : D2;

  const int tid  = threadIdx.x;
  const int lane = tid & 31;
  const int w    = tid >> 5;
  const int lr   = lane & 15;
  const int g    = lane >> 4;
  const int mBase = blockIdx.y * 128;
  const int nBase = blockIdx.x * 128;
  const int mtB = (w & 3) * 32;   // wave's 32 M rows
  const int ntB = (w >> 2) * 64;  // wave's 64 N cols

  v8f acc[2][4] = {{{}, {}, {}, {}}, {{}, {}, {}, {}}};

  for (int kb = 0; kb < EMBED; kb += 64) {
    // Stage X tile 128x64 halfs and W tile 128x64 halfs (both row-major)
#pragma unroll
    for (int j = 0; j < 4; ++j) {
      const int i = tid + 256 * j;  // 1024 16B chunks per matrix
      const int r = i >> 3, c8 = (i & 7) * 8;
      *(float4*)(lA + r * ASTR + c8) =
          *(const float4*)(X + (size_t)(mBase + r) * EMBED + kb + c8);
      *(float4*)(lB + r * ASTR + c8) =
          *(const float4*)(W + (size_t)(nBase + r) * EMBED + kb + c8);
    }
    __syncthreads();

#pragma unroll
    for (int kk = 0; kk < 2; ++kk) {  // two 32-wide K steps
      Frag16 a[2];
#pragma unroll
      for (int mi = 0; mi < 2; ++mi) {
        const half_t* ar = lA + (mtB + mi * 16 + lr) * ASTR + 32 * kk;
        a[mi].p.lo = *(const v8h*)(ar + 16 * g);       // K = 16g..16g+7
        a[mi].p.hi = *(const v8h*)(ar + 16 * g + 8);  // K = 16g+8..16g+15
      }
#pragma unroll
      for (int ni = 0; ni < 4; ++ni) {
        const half_t* br = lB + (ntB + ni * 16 + lr) * ASTR + 32 * kk + 16 * g;
        Frag16 b;
        b.p.lo = *(const v8h*)(br);      // K = 16g..16g+7
        b.p.hi = *(const v8h*)(br + 8);  // K = 16g+8..16g+15
#pragma unroll
        for (int mi = 0; mi < 2; ++mi)
          acc[mi][ni] = wmma32(a[mi].v, b.v, acc[mi][ni]);
      }
    }
    __syncthreads();
  }

#pragma unroll
  for (int ni = 0; ni < 4; ++ni) {
    const int n = nBase + ntB + ni * 16 + lr;
    const float bi = bias[n];
#pragma unroll
    for (int mi = 0; mi < 2; ++mi) {
#pragma unroll
      for (int r = 0; r < 8; ++r) {
        const int m = mBase + mtB + mi * 16 + r + 8 * g;
        const float val = acc[mi][ni][r] + bi;
        if (mode == 0) {
          const int bb = m >> 11, s = m & (SEQ - 1);
          Dh[(((size_t)bb * NH + (n >> 6)) * SEQ + s) * HD + (n & 63)] =
              (half_t)val;
        } else {
          Df[(size_t)m * EMBED + n] = val;
        }
      }
    }
  }
}

// ---------------------------------------------------------------------------
// Per-segment attention. grid=(B*NH*NSEG, 2); each wave owns one 16-row M tile.
// Q,K,V f16 in (B,H,S,D); ctx f16 in (B,S,E). Softmax exact, f32.
// ---------------------------------------------------------------------------
__global__ __launch_bounds__(256) void attn_kernel(
    const half_t* __restrict__ q, const half_t* __restrict__ k,
    const half_t* __restrict__ v, half_t* __restrict__ ctx) {
  extern __shared__ half_t smem[];
  half_t* lK  = smem;                    // 256 x KSTR
  half_t* lVT = smem + 256 * KSTR;       // 64 x VSTR (V transposed: [d][s])
  half_t* stg = lVT + 64 * VSTR;         // 8 waves x 16 x SSTR (P staging)

  int t = blockIdx.x;
  const int seg = t % NSEG; t /= NSEG;
  const int h = t % NH;
  const int b = t / NH;

  const int tid  = threadIdx.x;
  const int lane = tid & 31;
  const int w    = tid >> 5;
  const int lr   = lane & 15;
  const int g    = lane >> 4;

  const size_t base = (((size_t)b * NH + h) * SEQ + (size_t)seg * SEG) * HD;

  // Stage K (row-major, padded) and V (transposed [d][s]) segments
#pragma unroll
  for (int j = 0; j < 8; ++j) {
    const int i = tid + 256 * j;  // 2048 8-half chunks
    const int s = i >> 3, c8 = (i & 7) * 8;
    *(float4*)(lK + s * KSTR + c8) =
        *(const float4*)(k + base + (size_t)s * HD + c8);
    const v8h vv = *(const v8h*)(v + base + (size_t)s * HD + c8);
#pragma unroll
    for (int c = 0; c < 8; ++c) lVT[(c8 + c) * VSTR + s] = vv[c];
  }
  __syncthreads();

  const int mt = blockIdx.y * 8 + w;  // 16 M tiles of 16 rows
  const half_t* qrow = q + base + (size_t)(mt * 16 + lr) * HD;

  // Q A-fragments for K=64 (two 32-wide steps)
  Frag16 qa[2];
#pragma unroll
  for (int kk = 0; kk < 2; ++kk) {
    qa[kk].p.lo = *(const v8h*)(qrow + 32 * kk + 16 * g);
    qa[kk].p.hi = *(const v8h*)(qrow + 32 * kk + 16 * g + 8);
  }

  // Scores band: 16 x 256 (un-scaled; 1/sqrt(D) folded into expf)
  v8f sf[16];
#pragma unroll
  for (int nt = 0; nt < 16; ++nt) {
    v8f a = {};
#pragma unroll
    for (int kk = 0; kk < 2; ++kk) {
      const half_t* kr = lK + (nt * 16 + lr) * KSTR + 32 * kk + 16 * g;
      Frag16 bk;
      bk.p.lo = *(const v8h*)(kr);
      bk.p.hi = *(const v8h*)(kr + 8);
      a = wmma32(qa[kk].v, bk.v, a);
    }
    sf[nt] = a;
  }

  // Exact row softmax: row r+8g spans the 16 lanes of half g
  float rmax[8], inv[8];
#pragma unroll
  for (int r = 0; r < 8; ++r) {
    float mv = sf[0][r];
#pragma unroll
    for (int nt = 1; nt < 16; ++nt) mv = fmaxf(mv, sf[nt][r]);
#pragma unroll
    for (int off = 8; off >= 1; off >>= 1)
      mv = fmaxf(mv, __shfl_xor(mv, off, 32));
    rmax[r] = mv;
  }
#pragma unroll
  for (int r = 0; r < 8; ++r) {
    float s = 0.0f;
#pragma unroll
    for (int nt = 0; nt < 16; ++nt) {
      const float e = __expf((sf[nt][r] - rmax[r]) * 0.125f);
      sf[nt][r] = e;
      s += e;
    }
#pragma unroll
    for (int off = 8; off >= 1; off >>= 1) s += __shfl_xor(s, off, 32);
    inv[r] = 1.0f / s;
  }

  // ctx = P * V : stage P as f16 16x32 tiles (D-frag -> A-frag), V^T B-frags
  half_t* sw = stg + w * 16 * SSTR;
  v8f acc[4] = {{}, {}, {}, {}};
#pragma unroll
  for (int nt2 = 0; nt2 < 8; ++nt2) {  // 32-wide K slices of the 256 K dim
#pragma unroll
    for (int j2 = 0; j2 < 2; ++j2) {
      const int nt = nt2 * 2 + j2;
#pragma unroll
      for (int r = 0; r < 8; ++r)
        sw[(r + 8 * g) * SSTR + j2 * 16 + lr] = (half_t)sf[nt][r];
    }
    asm volatile("s_wait_dscnt 0" ::: "memory");
    Frag16 pa;
    pa.p.lo = *(const v8h*)(sw + lr * SSTR + 16 * g);
    pa.p.hi = *(const v8h*)(sw + lr * SSTR + 16 * g + 8);
#pragma unroll
    for (int dt = 0; dt < 4; ++dt) {
      const half_t* vr = lVT + (dt * 16 + lr) * VSTR + nt2 * 32 + 16 * g;
      Frag16 bv;
      bv.p.lo = *(const v8h*)(vr);
      bv.p.hi = *(const v8h*)(vr + 8);
      acc[dt] = wmma32(pa.v, bv.v, acc[dt]);
    }
  }

  // Normalize and store ctx (f16, (B,S,E) layout)
  const size_t orow0 = (size_t)b * SEQ + (size_t)seg * SEG + mt * 16;
#pragma unroll
  for (int dt = 0; dt < 4; ++dt) {
#pragma unroll
    for (int r = 0; r < 8; ++r) {
      const float val = acc[dt][r] * inv[r];
      ctx[(orow0 + r + 8 * g) * EMBED + h * HD + dt * 16 + lr] = (half_t)val;
    }
  }
}

extern "C" void kernel_launch(void* const* d_in, const int* in_sizes, int n_in,
                              void* d_out, int out_size, void* d_ws,
                              size_t ws_size, hipStream_t stream) {
  const float* x  = (const float*)d_in[0];
  const float* Wq = (const float*)d_in[1];
  const float* bq = (const float*)d_in[2];
  const float* Wk = (const float*)d_in[3];
  const float* bk = (const float*)d_in[4];
  const float* Wv = (const float*)d_in[5];
  const float* bv = (const float*)d_in[6];
  const float* Wo = (const float*)d_in[7];
  const float* bo = (const float*)d_in[8];
  float* out = (float*)d_out;

  const size_t nX = (size_t)MTOT * EMBED;    // 3,145,728
  const size_t nW = (size_t)EMBED * EMBED;   // 589,824
  const size_t nP = (size_t)BATCH * NH * SEQ * HD;  // == nX

  half_t* x_h  = (half_t*)d_ws;
  half_t* Wq_h = x_h + nX;
  half_t* Wk_h = Wq_h + nW;
  half_t* Wv_h = Wk_h + nW;
  half_t* Wo_h = Wv_h + nW;
  half_t* q_h  = Wo_h + nW;
  half_t* k_h  = q_h + nP;
  half_t* v_h  = k_h + nP;
  half_t* ctx_h = v_h + nP;  // total ~36 MB

  // 1) f32 -> f16 conversions
  cvt_f32_f16<<<dim3(nX / 4 / 256), 256, 0, stream>>>(x, x_h, (int)(nX / 4));
  cvt_f32_f16<<<dim3(nW / 4 / 256), 256, 0, stream>>>(Wq, Wq_h, (int)(nW / 4));
  cvt_f32_f16<<<dim3(nW / 4 / 256), 256, 0, stream>>>(Wk, Wk_h, (int)(nW / 4));
  cvt_f32_f16<<<dim3(nW / 4 / 256), 256, 0, stream>>>(Wv, Wv_h, (int)(nW / 4));
  cvt_f32_f16<<<dim3(nW / 4 / 256), 256, 0, stream>>>(Wo, Wo_h, (int)(nW / 4));

  // 2) QKV projections (f16 WMMA, f32 accumulate, f16 (B,H,S,D) outputs)
  gemm_f16_kernel<<<dim3(EMBED / 128, MTOT / 128, 3), 256, 0, stream>>>(
      x_h, Wq_h, Wk_h, Wv_h, bq, bk, bv, q_h, k_h, v_h, nullptr, 0);

  // 3) Block-diagonal attention
  const size_t shmem =
      (size_t)(256 * KSTR + 64 * VSTR + 8 * 16 * SSTR) * sizeof(half_t);
  attn_kernel<<<dim3(BATCH * NH * NSEG, 2), 256, shmem, stream>>>(q_h, k_h,
                                                                  v_h, ctx_h);

  // 4) Output projection (f32 result + bias)
  gemm_f16_kernel<<<dim3(EMBED / 128, MTOT / 128, 1), 256, 0, stream>>>(
      ctx_h, Wo_h, Wo_h, Wo_h, bo, bo, bo, q_h, q_h, q_h, out, 1);
}
